// StreamingEncoderV2_45999099740717
// MI455X (gfx1250) — matrix operand order repaired and checked by reading, not compiled, not run on hardware
//
#include <hip/hip_runtime.h>
#include <hip/hip_bf16.h>
#include <utility>

// ---------------- constants ----------------
namespace se {
constexpr int B = 32;
constexpr int FRAME = 1920;
constexpr int DM = 512;
constexpr int NH = 8;
constexpr int HD = 64;
constexpr int NL = 8;
constexpr int FFNDIM = 2048;
constexpr int PAST = 500;
constexpr int TT = 2;              // new tokens per batch element
constexpr int TOT = PAST + TT;     // 502
constexpr int NQ = 8;
constexpr int CBS = 2048;
constexpr int CBD = 256;
}

typedef _Float16 f16;
typedef __attribute__((ext_vector_type(4)))  _Float16 v4h;
typedef __attribute__((ext_vector_type(16))) _Float16 v16h;
typedef __attribute__((ext_vector_type(8)))  float    v8f;
typedef __attribute__((ext_vector_type(4)))  float    v4f;   // native vec for NT builtins

// ---------------- elementwise ----------------
__global__ void k_elu(const float* __restrict__ x, float* __restrict__ y, long n) {
  long i = (long)blockIdx.x * blockDim.x + threadIdx.x;
  if (i >= n) return;
  float v = x[i];
  y[i] = v > 0.f ? v : (expf(v) - 1.f);
}

__global__ void k_gelu(float* __restrict__ x, long n) {
  long i = (long)blockIdx.x * blockDim.x + threadIdx.x;
  if (i >= n) return;
  float v = x[i];
  x[i] = 0.5f * v * (1.f + erff(v * 0.70710678118654752f));
}

__global__ void k_add(float* __restrict__ dst, const float* __restrict__ src, long n) {
  long i = (long)blockIdx.x * blockDim.x + threadIdx.x;
  if (i >= n) return;
  dst[i] += src[i];
}

__global__ void k_scale_add(float* __restrict__ h, const float* __restrict__ y,
                            const float* __restrict__ s, long n, int D) {
  long i = (long)blockIdx.x * blockDim.x + threadIdx.x;
  if (i >= n) return;
  h[i] += s[i % D] * y[i];
}

// vectorized f32 -> f16 convert: b128 load, b64 store (n must be multiple of 4)
__global__ void k_cvt_f16_v4(const v4f* __restrict__ x, f16* __restrict__ y, long n4) {
  long i = (long)blockIdx.x * blockDim.x + threadIdx.x;
  if (i >= n4) return;
  v4f v = x[i];
  v4h o;
  o[0] = (f16)v[0]; o[1] = (f16)v[1]; o[2] = (f16)v[2]; o[3] = (f16)v[3];
  *(v4h*)(y + i * 4) = o;
}

// ---------------- streaming conv helpers ----------------
__global__ void k_concat(const float* __restrict__ st, const float* __restrict__ x,
                         float* __restrict__ xc, int rows, int S, int T) {
  int Wd = S + T;
  long tot = (long)rows * Wd;
  long i = (long)blockIdx.x * blockDim.x + threadIdx.x;
  if (i >= tot) return;
  int r = (int)(i / Wd), c = (int)(i % Wd);
  xc[i] = (c < S) ? st[(size_t)r * S + c] : x[(size_t)r * T + (c - S)];
}

// new_state = last S columns of x (T >= S for every stage here)
__global__ void k_tail(const float* __restrict__ x, float* __restrict__ out,
                       int rows, int T, int S) {
  long tot = (long)rows * S;
  long i = (long)blockIdx.x * blockDim.x + threadIdx.x;
  if (i >= tot) return;
  int r = (int)(i / S), s = (int)(i % S);
  out[i] = x[(size_t)r * T + (T - S) + s];
}

// plain VALID conv1d, dilation=1
__global__ void k_conv1d(const float* __restrict__ xc, const float* __restrict__ w,
                         const float* __restrict__ bias, float* __restrict__ y,
                         int Cin, int Cout, int Win, int Tout, int K, int stride) {
  long tot = (long)se::B * Cout * Tout;
  long i = (long)blockIdx.x * blockDim.x + threadIdx.x;
  if (i >= tot) return;
  int t  = (int)(i % Tout);
  int co = (int)((i / Tout) % Cout);
  int b  = (int)(i / ((long)Tout * Cout));
  float acc = bias[co];
  const float* xb = xc + (size_t)b * Cin * Win + (size_t)t * stride;
  const float* wb = w + (size_t)co * Cin * K;
  for (int ci = 0; ci < Cin; ++ci) {
    const float* xr = xb + (size_t)ci * Win;
    const float* wr = wb + (size_t)ci * K;
    __builtin_prefetch(xr + Win, 0, 1);   // -> global_prefetch_b8 (next channel row)
    for (int k = 0; k < K; ++k) acc += xr[k] * wr[k];
  }
  y[i] = acc;
}

// ---------------- layout shuffles ----------------
__global__ void k_chw_to_tok(const float* __restrict__ x, float* __restrict__ h) {
  long n = (long)se::B * se::TT * se::DM;
  long i = (long)blockIdx.x * blockDim.x + threadIdx.x;
  if (i >= n) return;
  int d = (int)(i % se::DM);
  int bt = (int)(i / se::DM);
  int t = bt % se::TT, b = bt / se::TT;
  h[i] = x[((size_t)b * se::DM + d) * se::TT + t];
}

__global__ void k_tok_to_chw(const float* __restrict__ h, float* __restrict__ x) {
  long n = (long)se::B * se::DM * se::TT;
  long i = (long)blockIdx.x * blockDim.x + threadIdx.x;
  if (i >= n) return;
  int t = (int)(i % se::TT);
  int d = (int)((i / se::TT) % se::DM);
  int b = (int)(i / ((long)se::TT * se::DM));
  x[i] = h[((size_t)(b * se::TT + t)) * se::DM + d];
}

// ---------------- layernorm (one block per row) ----------------
__global__ void k_layernorm(const float* __restrict__ x, const float* __restrict__ g,
                            const float* __restrict__ bt, float* __restrict__ y, int D) {
  int row = blockIdx.x, tid = threadIdx.x;  // 256 threads
  __shared__ float r1[256], r2[256];
  const float* xr = x + (size_t)row * D;
  float s = 0.f, s2 = 0.f;
  for (int d = tid; d < D; d += 256) { float v = xr[d]; s += v; s2 += v * v; }
  r1[tid] = s; r2[tid] = s2; __syncthreads();
  for (int st = 128; st > 0; st >>= 1) {
    if (tid < st) { r1[tid] += r1[tid + st]; r2[tid] += r2[tid + st]; }
    __syncthreads();
  }
  float mean = r1[0] / D;
  float var  = r2[0] / D - mean * mean;
  float inv  = rsqrtf(var + 1e-5f);
  for (int d = tid; d < D; d += 256)
    y[(size_t)row * D + d] = (xr[d] - mean) * inv * g[d] + bt[d];
}

// ---------------- WMMA GEMM: Y[M,N] = Xh[M,K] @ Wh[N,K]^T (+bias) ----------------
__global__ void k_gemm_wmma(const f16* __restrict__ X, const f16* __restrict__ W,
                            const float* __restrict__ bias, float* __restrict__ Y,
                            int M, int K, int N) {
  int wave = (int)((blockIdx.x * blockDim.x + threadIdx.x) >> 5);
  int lane = threadIdx.x & 31;
  int tn = N >> 4;
  int mt = wave / tn;
  int nt = wave - mt * tn;
  if (mt >= (M >> 4)) return;             // wave-uniform exit; EXEC stays all-ones
  int lo = lane & 15;
  int hi = lane >> 4;                      // 0 or 1
  const f16* xrow = X + (size_t)(mt * 16 + lo) * K;   // A: lane -> row M
  const f16* wrow = W + (size_t)(nt * 16 + lo) * K;   // B: lane -> col N
  v8f acc = {};
  for (int k0 = 0; k0 < K; k0 += 32) {
    v16h a, b;
    const f16* xa = xrow + k0 + hi * 8;    // lanes 0-15: K {0..7,16..23}; 16-31: {8..15,24..31}
    const f16* wb = wrow + k0 + hi * 8;    // lanes 0-15: K {0..7,16..23}; 16-31: {8..15,24..31}
#pragma unroll
    for (int j = 0; j < 8; ++j) { a[j] = xa[j]; a[j + 8] = xa[j + 16]; }
#pragma unroll
    for (int j = 0; j < 8; ++j) { b[j] = wb[j]; b[j + 8] = wb[j + 16]; }
    acc = __builtin_amdgcn_wmma_f32_16x16x32_f16(
        /*neg_a=*/false, a, /*neg_b=*/false, b,
        /*c_mod=*/(short)0, acc, /*reuse_a=*/false, /*reuse_b=*/false);
  }
  int col = nt * 16 + lo;
  float bv = bias ? bias[col] : 0.f;
  float* yb = Y + (size_t)(mt * 16 + hi * 8) * N + col;   // C/D: VGPR r -> row r (+8 for hi)
#pragma unroll
  for (int r = 0; r < 8; ++r) yb[(size_t)r * N] = acc[r] + bv;
}

// ---------------- RoPE + scatter new K/V into the concatenated cache ----------------
__global__ void k_rope_write(const float* __restrict__ qb, const float* __restrict__ kb,
                             const float* __restrict__ vb, float* __restrict__ qo,
                             float* __restrict__ ko, float* __restrict__ vo) {
  int n = se::B * se::TT * se::DM;
  int i = blockIdx.x * blockDim.x + threadIdx.x;
  if (i >= n) return;
  int d = i & (se::HD - 1);
  int h = (i >> 6) & (se::NH - 1);
  int bt = i >> 9;
  int t = bt % se::TT, b = bt / se::TT;
  float pos = (float)(se::PAST + t);
  float invf = powf(10000.f, -(float)(d & 31) / 32.f);
  float ang = pos * invf;
  float c = cosf(ang), s = sinf(ang);
  float qr = (d < 32) ? -qb[i + 32] : qb[i - 32];
  float kr = (d < 32) ? -kb[i + 32] : kb[i - 32];
  qo[i] = qb[i] * c + qr * s;
  size_t oidx = ((size_t)(b * se::NH + h) * se::TOT + se::PAST + t) * se::HD + d;
  ko[oidx] = kb[i] * c + kr * s;
  vo[oidx] = vb[i];
}

// old KV (B,H,500,64) -> out (B,H,502,64), rows 0..499.
// 16B-per-lane non-temporal copy: streaming data, keep it out of L2 as much as possible.
__global__ void k_kvcopy4(const v4f* __restrict__ in, v4f* __restrict__ out, int n4) {
  int i = blockIdx.x * blockDim.x + threadIdx.x;
  if (i >= n4) return;
  int d = i & 15;                // 16 float4 per 64-elem row
  int rest = i >> 4;
  int t = rest % se::PAST;
  int bh = rest / se::PAST;
  v4f v = __builtin_nontemporal_load(in + i);
  __builtin_nontemporal_store(v, out + ((size_t)bh * se::TOT + t) * 16 + d);
}

// Same copy via the CDNA5 async-LDS DMA path (ASYNCcnt):
// global_load_async_to_lds_b128 -> s_wait_asynccnt -> global_store_async_from_lds_b128.
// Bypasses VGPRs entirely; each lane owns a private 16B LDS slot.
__global__ void k_kvcopy_async(const v4f* __restrict__ in, v4f* __restrict__ out, int n4) {
  __shared__ alignas(16) v4f stage[256];
  int i = blockIdx.x * blockDim.x + threadIdx.x;
  if (i >= n4) return;
  int d = i & 15;
  int rest = i >> 4;
  int t = rest % se::PAST;
  int bh = rest / se::PAST;
  const v4f* src = in + i;
  v4f* dst = out + ((size_t)bh * se::TOT + t) * 16 + d;
  unsigned lds = (unsigned)(size_t)&stage[threadIdx.x];   // low 32 bits = LDS byte offset
  asm volatile("global_load_async_to_lds_b128 %0, %1, off"
               :: "v"(lds), "v"(src) : "memory");
  asm volatile("s_wait_asynccnt 0x0" ::: "memory");
  asm volatile("global_store_async_from_lds_b128 %0, %1, off"
               :: "v"(dst), "v"(lds) : "memory");
  asm volatile("s_wait_asynccnt 0x0" ::: "memory");
}

// ---------------- attention: one block per (b,h,t) ----------------
__global__ void k_attn(const float* __restrict__ q, const float* __restrict__ kc,
                       const float* __restrict__ vc, float* __restrict__ o) {
  int bid = blockIdx.x;                 // ((b*NH + h)*TT + t)
  int t = bid % se::TT;
  int bh = bid / se::TT;
  int h = bh % se::NH;
  int b = bh / se::NH;
  int tid = threadIdx.x;                // 128 threads
  __shared__ alignas(16) float qs[se::HD];
  __shared__ float sc[se::TOT];
  __shared__ float red[128];
  const float* qp = q + ((size_t)(b * se::TT + t) * se::DM + h * se::HD);
  if (tid < se::HD) qs[tid] = qp[tid];
  __syncthreads();
  int keys = se::PAST + 1 + t;          // causal: 501 or 502
  const float* kb = kc + (size_t)bh * se::TOT * se::HD;
  const float* vb = vc + (size_t)bh * se::TOT * se::HD;
  const v4f* qs4 = reinterpret_cast<const v4f*>(qs);
  float mx = -1e30f;
  for (int k = tid; k < keys; k += 128) {
    const v4f* kr4 = reinterpret_cast<const v4f*>(kb + (size_t)k * se::HD);
    float d = 0.f;
#pragma unroll
    for (int j = 0; j < se::HD / 4; ++j) {
      v4f a = qs4[j], c = kr4[j];
      d += a[0] * c[0] + a[1] * c[1] + a[2] * c[2] + a[3] * c[3];
    }
    d *= 0.125f;                        // 1/sqrt(64)
    sc[k] = d;
    mx = fmaxf(mx, d);
  }
  red[tid] = mx; __syncthreads();
  for (int s = 64; s > 0; s >>= 1) { if (tid < s) red[tid] = fmaxf(red[tid], red[tid + s]); __syncthreads(); }
  mx = red[0]; __syncthreads();
  float sum = 0.f;
  for (int k = tid; k < keys; k += 128) { float e = expf(sc[k] - mx); sc[k] = e; sum += e; }
  red[tid] = sum; __syncthreads();
  for (int s = 64; s > 0; s >>= 1) { if (tid < s) red[tid] += red[tid + s]; __syncthreads(); }
  float inv = 1.f / red[0];
  __syncthreads();
  if (tid < se::HD) {
    float acc = 0.f;
    for (int k = 0; k < keys; ++k) acc += sc[k] * vb[(size_t)k * se::HD + tid];
    o[(size_t)(b * se::TT + t) * se::DM + h * se::HD + tid] = acc * inv;
  }
}

// ---------------- RVQ: per-row argmin over codebook (fp32 for accuracy) ----------------
__global__ void k_rvq_argmin(const float* __restrict__ r, const float* __restrict__ cb,
                             float* __restrict__ codes, int qi, int* __restrict__ idxout) {
  int row = blockIdx.x;                 // 0..31
  int tid = threadIdx.x;                // 256
  __shared__ float rs[se::CBD];
  __shared__ float bv[256];
  __shared__ int   bi[256];
  rs[tid] = r[(size_t)row * se::CBD + tid];
  __syncthreads();
  float best = 3.4e38f; int besti = 0;
  for (int j = tid; j < se::CBS; j += 256) {
    const float* c = cb + (size_t)j * se::CBD;
    float d = 0.f;
    for (int k = 0; k < se::CBD; ++k) { float df = rs[k] - c[k]; d += df * df; }
    if (d < best || (d == best && j < besti)) { best = d; besti = j; }
  }
  bv[tid] = best; bi[tid] = besti; __syncthreads();
  for (int s = 128; s > 0; s >>= 1) {
    if (tid < s) {
      if (bv[tid + s] < bv[tid] || (bv[tid + s] == bv[tid] && bi[tid + s] < bi[tid])) {
        bv[tid] = bv[tid + s]; bi[tid] = bi[tid + s];
      }
    }
    __syncthreads();
  }
  if (tid == 0) {
    codes[row * se::NQ + qi] = (float)bi[0];
    idxout[row] = bi[0];
  }
}

__global__ void k_rvq_sub(float* __restrict__ r, const float* __restrict__ cb,
                          const int* __restrict__ idx) {
  int i = blockIdx.x * blockDim.x + threadIdx.x;
  if (i >= se::B * se::CBD) return;
  int row = i / se::CBD, d = i % se::CBD;
  r[i] -= cb[(size_t)idx[row] * se::CBD + d];
}

// ---------------- host orchestration ----------------
extern "C" void kernel_launch(void* const* d_in, const int* in_sizes, int n_in,
                              void* d_out, int out_size, void* d_ws, size_t ws_size,
                              hipStream_t stream) {
  (void)in_sizes; (void)n_in; (void)out_size; (void)ws_size;
  using namespace se;

  auto IN = [&](int i) { return (const float*)d_in[i]; };
  const float* x_in = IN(0);
  const float* cstate[11];
  for (int i = 0; i < 11; ++i) cstate[i] = IN(1 + i);
  const float* kvk[8]; const float* kvv[8];
  for (int l = 0; l < 8; ++l) { kvk[l] = IN(12 + 2 * l); kvv[l] = IN(13 + 2 * l); }
  auto P = [&](int i) { return (const float*)d_in[28 + i]; };  // params in dict insertion order

  // ---- output offsets: codes, states 0..10, (k,v) x 8 layers ----
  static const int st_c[11] = {1, 64, 64, 128, 128, 256, 256, 512, 512, 1024, 512};
  static const int st_t[11] = {6, 2, 4, 2, 5, 2, 6, 2, 8, 2, 2};
  float* out = (float*)d_out;
  size_t oo = 0;
  float* codes = out; oo += (size_t)B * NQ;
  float* ostate[11];
  for (int i = 0; i < 11; ++i) { ostate[i] = out + oo; oo += (size_t)B * st_c[i] * st_t[i]; }
  float* okv[16];
  for (int l = 0; l < 8; ++l) {
    okv[2 * l]     = out + oo; oo += (size_t)B * NH * TOT * HD;
    okv[2 * l + 1] = out + oo; oo += (size_t)B * NH * TOT * HD;
  }

  // ---- workspace carve (all offsets stay 16B aligned) ----
  float* WSP = (float*)d_ws;
  size_t w = 0;
  auto alloc = [&](size_t n) { float* p = WSP + w; w += n; return p; };
  const size_t BIG = 4000000;       // covers (B,64,1922) = 3,936,256 floats
  float* ACT = alloc(BIG);
  float* TMP = alloc(BIG);
  float* XC  = alloc(BIG);
  float* h   = alloc(64 * 512);
  float* hn  = alloc(64 * 512);
  float* qb  = alloc(64 * 512);
  float* kb  = alloc(64 * 512);
  float* vb  = alloc(64 * 512);
  float* qro = alloc(64 * 512);
  float* ob  = alloc(64 * 512);
  float* mid = alloc(64 * 2048);
  float* yb  = alloc(64 * 2048);
  f16* xh = (f16*)alloc(64 * 2048 / 2 + 16);           // 131072 halves
  f16* wh = (f16*)alloc((size_t)2048 * 512 / 2 + 16);  // 1,048,576 halves
  float* rbuf = alloc(32 * 256);
  int* idxb = (int*)alloc(64);

  auto g1 = [&](long n) { return dim3((unsigned)((n + 255) / 256)); };
  auto gemm = [&](const f16* X, const f16* Wt, const float* bias, float* Y,
                  int M, int K, int N) {
    int tiles = (M / 16) * (N / 16);
    k_gemm_wmma<<<dim3((unsigned)((tiles + 7) / 8)), 256, 0, stream>>>(X, Wt, bias, Y, M, K, N);
  };
  auto cvt = [&](const float* src, f16* dst, long n) {
    k_cvt_f16_v4<<<g1(n / 4), 256, 0, stream>>>((const v4f*)src, dst, n / 4);
  };

  // ---- SEANet encoder ----
  int T = FRAME;
  auto do_conv = [&](const float* src, int sidx, const float* wgt, const float* bias,
                     int Cin, int Cout, int K, int S) {
    int Slen = st_t[sidx];
    int rows = B * Cin;
    int Win = Slen + T;
    k_concat<<<g1((long)rows * Win), 256, 0, stream>>>(cstate[sidx], src, XC, rows, Slen, T);
    k_tail<<<g1((long)rows * Slen), 256, 0, stream>>>(src, ostate[sidx], rows, T, Slen);
    int Tout = (Win + S - K) / S;
    k_conv1d<<<g1((long)B * Cout * Tout), 256, 0, stream>>>(XC, wgt, bias, TMP, Cin, Cout, Win, Tout, K, S);
    std::swap(ACT, TMP);
    T = Tout;
  };
  auto do_res = [&](int sidx, const float* aw, const float* ab, const float* bw,
                    const float* bb, int C) {
    long n = (long)B * C * T;
    k_elu<<<g1(n), 256, 0, stream>>>(ACT, TMP, n);                        // h = elu(x)
    int rows = B * C, Slen = st_t[sidx], Win = Slen + T;
    k_concat<<<g1((long)rows * Win), 256, 0, stream>>>(cstate[sidx], TMP, XC, rows, Slen, T);
    k_tail<<<g1((long)rows * Slen), 256, 0, stream>>>(TMP, ostate[sidx], rows, T, Slen);
    k_conv1d<<<g1((long)B * (C / 2) * T), 256, 0, stream>>>(XC, aw, ab, TMP, C, C / 2, Win, T, 3, 1);
    k_elu<<<g1((long)B * (C / 2) * T), 256, 0, stream>>>(TMP, TMP, (long)B * (C / 2) * T);
    k_conv1d<<<g1(n), 256, 0, stream>>>(TMP, bw, bb, XC, C / 2, C, T, T, 1, 1);
    k_add<<<g1(n), 256, 0, stream>>>(ACT, XC, n);                         // x = x + r
  };
  auto do_elu = [&](int C) {
    long n = (long)B * C * T;
    k_elu<<<g1(n), 256, 0, stream>>>(ACT, ACT, n);
  };

  do_conv(x_in, 0, P(0), P(1), 1, 64, 7, 1);          // enc0
  do_res(1, P(2), P(3), P(4), P(5), 64);              // enc1
  do_elu(64);
  do_conv(ACT, 2, P(6), P(7), 64, 128, 8, 4);         // enc3 -> T=480
  do_res(3, P(8), P(9), P(10), P(11), 128);           // enc4
  do_elu(128);
  do_conv(ACT, 4, P(12), P(13), 128, 256, 10, 5);     // enc6 -> T=96
  do_res(5, P(14), P(15), P(16), P(17), 256);         // enc7
  do_elu(256);
  do_conv(ACT, 6, P(18), P(19), 256, 512, 12, 6);     // enc9 -> T=16
  do_res(7, P(20), P(21), P(22), P(23), 512);         // enc10
  do_elu(512);
  do_conv(ACT, 8, P(24), P(25), 512, 1024, 16, 8);    // enc12 -> T=2
  do_elu(1024);
  do_conv(ACT, 9, P(26), P(27), 1024, 512, 3, 1);     // enc14 -> (B,512,2)

  // ---- transformer ----
  k_chw_to_tok<<<g1(64 * 512), 256, 0, stream>>>(ACT, h);
  int nkv4 = B * NH * PAST * HD / 4;                  // float4 units per KV tensor
  for (int l = 0; l < NL; ++l) {
    int pb = 28 + 14 * l;
    const float *wq = P(pb + 0), *wk_ = P(pb + 1), *wv = P(pb + 2), *wo = P(pb + 3);
    const float *g1g = P(pb + 4), *b1 = P(pb + 5), *g2 = P(pb + 6), *b2 = P(pb + 7);
    const float *f1w = P(pb + 8), *f1b = P(pb + 9), *f2w = P(pb + 10), *f2b = P(pb + 11);
    const float *s1 = P(pb + 12), *s2 = P(pb + 13);

    k_layernorm<<<64, 256, 0, stream>>>(h, g1g, b1, hn, 512);
    cvt(hn, xh, 64 * 512);
    cvt(wq, wh, 512 * 512);  gemm(xh, wh, nullptr, qb, 64, 512, 512);
    cvt(wk_, wh, 512 * 512); gemm(xh, wh, nullptr, kb, 64, 512, 512);
    cvt(wv, wh, 512 * 512);  gemm(xh, wh, nullptr, vb, 64, 512, 512);

    // K cache via vectorized non-temporal VGPR path
    k_kvcopy4<<<g1(nkv4), 256, 0, stream>>>((const v4f*)kvk[l], (v4f*)okv[2 * l], nkv4);
    // V cache via async-LDS DMA path (ASYNCcnt)
    k_kvcopy_async<<<g1(nkv4), 256, 0, stream>>>((const v4f*)kvv[l], (v4f*)okv[2 * l + 1], nkv4);

    k_rope_write<<<g1(64 * 512), 256, 0, stream>>>(qb, kb, vb, qro, okv[2 * l], okv[2 * l + 1]);
    k_attn<<<B * NH * TT, 128, 0, stream>>>(qro, okv[2 * l], okv[2 * l + 1], ob);

    cvt(ob, xh, 64 * 512);
    cvt(wo, wh, 512 * 512);  gemm(xh, wh, nullptr, yb, 64, 512, 512);
    k_scale_add<<<g1(64 * 512), 256, 0, stream>>>(h, yb, s1, 64 * 512, 512);

    k_layernorm<<<64, 256, 0, stream>>>(h, g2, b2, hn, 512);
    cvt(hn, xh, 64 * 512);
    cvt(f1w, wh, (long)2048 * 512); gemm(xh, wh, f1b, mid, 64, 512, 2048);
    k_gelu<<<g1(64 * 2048), 256, 0, stream>>>(mid, 64 * 2048);
    cvt(mid, xh, 64 * 2048);
    cvt(f2w, wh, (long)512 * 2048); gemm(xh, wh, f2b, yb, 64, 2048, 512);
    k_scale_add<<<g1(64 * 512), 256, 0, stream>>>(h, yb, s2, 64 * 512, 512);
  }

  // ---- downsample conv (state 10) ----
  k_tok_to_chw<<<g1(64 * 512), 256, 0, stream>>>(h, ACT);           // (B,512,2)
  {
    int rows = B * 512;
    k_concat<<<g1((long)rows * 4), 256, 0, stream>>>(cstate[10], ACT, XC, rows, 2, 2);
    k_tail<<<g1((long)rows * 2), 256, 0, stream>>>(ACT, ostate[10], rows, 2, 2);
    k_conv1d<<<g1((long)B * 512 * 1), 256, 0, stream>>>(XC, P(140), P(141), TMP, 512, 512, 4, 1, 4, 2);
  }  // TMP = emb (B,512,1) == row-major (32,512)

  // ---- RVQ ----
  cvt(TMP, xh, 32 * 512);                               // emb in f16, reused for both projections
  cvt(P(142), wh, (long)256 * 512);                     // q_sem_proj
  gemm(xh, wh, nullptr, rbuf, 32, 512, 256);
  k_rvq_argmin<<<32, 256, 0, stream>>>(rbuf, P(144), codes, 0, idxb);

  cvt(P(143), wh, (long)256 * 512);                     // q_ac_proj
  gemm(xh, wh, nullptr, rbuf, 32, 512, 256);
  for (int i = 0; i < 7; ++i) {
    const float* cb = P(145) + (size_t)i * CBS * CBD;   // q_ac_cb[i]
    k_rvq_argmin<<<32, 256, 0, stream>>>(rbuf, cb, codes, 1 + i, idxb);
    k_rvq_sub<<<g1(32 * 256), 256, 0, stream>>>(rbuf, cb, idxb);
  }
}
